// EnhancedBEVModule_80882824119010
// MI455X (gfx1250) — compile-verified, matrix operand order checked by reading, not run on hardware
//
#include <hip/hip_runtime.h>

// ---------------------------------------------------------------------------
// Problem constants (from the reference)
// ---------------------------------------------------------------------------
#define BB 32
#define NN 200000
#define SS 6
#define HH 160
#define WW 160
#define PP 100          // patches per slice (10x10)
#define DD 256
#define BS (BB * SS)    // 192
#define MROWS (PP * BS) // 19200 token rows
#define EPSF 1e-6f

typedef _Float16 f16;
typedef _Float16 v16h __attribute__((ext_vector_type(16)));
typedef float    v8f  __attribute__((ext_vector_type(8)));

// ---------------------------------------------------------------------------
// Stage 1: per-batch z min/max -> slice edges
// ---------------------------------------------------------------------------
__global__ void k_zminmax(const float* __restrict__ xyz, float* __restrict__ edges) {
  __shared__ float smn[256], smx[256];
  const int b = blockIdx.x, t = threadIdx.x;
  float mn = 1e30f, mx = -1e30f;
  for (int n = t; n < NN; n += 256) {
    float z = xyz[((size_t)b * NN + n) * 3 + 2];
    mn = fminf(mn, z);
    mx = fmaxf(mx, z);
  }
  smn[t] = mn; smx[t] = mx;
  __syncthreads();
  for (int s = 128; s > 0; s >>= 1) {
    if (t < s) { smn[t] = fminf(smn[t], smn[t + s]); smx[t] = fmaxf(smx[t], smx[t + s]); }
    __syncthreads();
  }
  if (t == 0) {
    float zmin = smn[0], zmax = smx[0];
    for (int j = 0; j <= SS; ++j)
      edges[b * (SS + 1) + j] = zmin + (zmax - zmin) * ((float)j / (float)SS);
  }
}

// ---------------------------------------------------------------------------
// Stage 2: histogram scatter
// ---------------------------------------------------------------------------
__global__ void k_scatter(const float* __restrict__ xyz, const float* __restrict__ edges,
                          unsigned* __restrict__ counts) {
  size_t id = (size_t)blockIdx.x * 256 + threadIdx.x;
  if (id >= (size_t)BB * NN) return;
  int b = (int)(id / NN);
  const float* p = xyz + id * 3;
  float x = p[0], y = p[1], z = p[2];
  float gx = (x + 1.0f) / (2.0f + EPSF) * (float)(WW - 1);
  float gy = (y + 1.0f) / (2.0f + EPSF) * (float)(HH - 1);
  int sidx = 0;
#pragma unroll
  for (int s = 1; s <= SS; ++s) sidx += (z >= edges[b * (SS + 1) + s]) ? 1 : 0;
  bool valid = (gy >= 0.0f) && (gy < (float)HH) && (gx >= 0.0f) && (gx < (float)WW) && (sidx < SS);
  if (valid) {
    int iy = (int)gy; iy = iy < 0 ? 0 : (iy > HH - 1 ? HH - 1 : iy);
    int ix = (int)gx; ix = ix < 0 ? 0 : (ix > WW - 1 ? WW - 1 : ix);
    int flat = ((b * SS + sidx) * HH + iy) * WW + ix;
    atomicAdd(&counts[flat], 1u);
  }
}

// ---------------------------------------------------------------------------
// Stage 3: log1p in place + per-(b,s) min/max
// ---------------------------------------------------------------------------
__global__ void k_bevlog(unsigned* __restrict__ counts, float* __restrict__ sMn,
                         float* __restrict__ sMx) {
  __shared__ float smn[256], smx[256];
  const int bs = blockIdx.x, t = threadIdx.x;
  float* f = (float*)counts;
  float mn = 1e30f, mx = -1e30f;
  for (int i = t; i < HH * WW; i += 256) {
    size_t idx = (size_t)bs * (HH * WW) + i;
    float v = log1pf((float)counts[idx]);
    f[idx] = v;
    mn = fminf(mn, v);
    mx = fmaxf(mx, v);
  }
  smn[t] = mn; smx[t] = mx;
  __syncthreads();
  for (int s = 128; s > 0; s >>= 1) {
    if (t < s) { smn[t] = fminf(smn[t], smn[t + s]); smx[t] = fmaxf(smx[t], smx[t + s]); }
    __syncthreads();
  }
  if (t == 0) { sMn[bs] = smn[0]; sMx[bs] = smx[0]; }
}

// ---------------------------------------------------------------------------
// Stage 4: normalize + patchify -> f16 token matrix X[MROWS][256]
// ---------------------------------------------------------------------------
__global__ void k_patchify(const float* __restrict__ bevlog, const float* __restrict__ sMn,
                           const float* __restrict__ sMx, f16* __restrict__ X) {
  const int r = blockIdx.x, c = threadIdx.x;
  int p = r / BS, bs = r % BS;
  int py = p / 10, px = p % 10;
  int dy = c / 16, dx = c % 16;
  float v = bevlog[(size_t)bs * (HH * WW) + (py * 16 + dy) * WW + (px * 16 + dx)];
  float mn = sMn[bs], mx = sMx[bs];
  X[(size_t)r * DD + c] = (f16)((v - mn) / (mx - mn + EPSF));
}

// ---------------------------------------------------------------------------
// f32 -> f16 convert WITH transpose: s is [Kd x Nd] row-major, d is [Nd x Kd]
// (weights pre-transposed once so GEMM B-tiles are contiguous row segments)
// ---------------------------------------------------------------------------
__global__ void k_cvtT(const float* __restrict__ s, f16* __restrict__ d, int Kd, int Nd) {
  int i = blockIdx.x * 256 + threadIdx.x;
  if (i < Kd * Nd) {
    int k = i / Nd, n = i % Nd;
    d[(size_t)n * Kd + k] = (f16)s[i];
  }
}

// ---------------------------------------------------------------------------
// WMMA GEMM: C[M x N] = A[M x K](f16,row-major) * Bt[N x K](f16)^T + bias.
// Block = 256 threads = 8 waves; block tile 64(M) x 64(N); K-step 32.
// Waves: wm = w&3 (M subtile), wn = w>>2; each wave owns two 16x16 N-subtiles.
// Double-buffered LDS fed by global_load_async_to_lds_b128 (ASYNCcnt).
// OUTM: 0 = f32 only, 1 = f16 only, 2 = both.  OP: 0 none, 1 relu, 2 elu+1.
// ---------------------------------------------------------------------------
template <int K, int N, int OUTM, int OP>
__launch_bounds__(256)
__global__ void k_gemm(const f16* __restrict__ A, const f16* __restrict__ Bt,
                       const float* __restrict__ bias, float* __restrict__ Cf,
                       f16* __restrict__ Ch) {
  __shared__ f16 lA[2][64 * 32];
  __shared__ f16 lB[2][64 * 32];
  const unsigned t = threadIdx.x;
  const unsigned w = t >> 5, lane = t & 31;
  const unsigned wm = w & 3, wn = w >> 2;
  const unsigned row0 = blockIdx.x * 64;
  const unsigned col0 = blockIdx.y * 64;
  // staging assignment: thread -> (row, 16B segment) for both A and Bt tiles
  const unsigned lr = t >> 2;          // 0..63
  const unsigned lseg = (t & 3) * 8;   // half offset (8 halves = 16 B)
  const f16* gA = A + (size_t)(row0 + lr) * K + lseg;
  const f16* gB = Bt + (size_t)(col0 + lr) * K + lseg;

  auto issue = [&](int k0, int buf) {
    unsigned dA = (unsigned)(size_t)(&lA[buf][lr * 32 + lseg]);
    unsigned dB = (unsigned)(size_t)(&lB[buf][lr * 32 + lseg]);
    const f16* sA = gA + k0;
    const f16* sB = gB + k0;
    asm volatile("global_load_async_to_lds_b128 %0, %1, off"
                 :: "v"(dA), "v"(sA) : "memory");
    asm volatile("global_load_async_to_lds_b128 %0, %1, off"
                 :: "v"(dB), "v"(sB) : "memory");
  };

  v8f c0 = {}, c1 = {};
  constexpr int NT = K / 32;
  issue(0, 0);
  asm volatile("s_wait_asynccnt 0x0" ::: "memory");
  __syncthreads();
  for (int kt = 0; kt < NT; ++kt) {
    const int cur = kt & 1;
    if (kt + 1 < NT) issue((kt + 1) * 32, 1 - cur);  // prefetch next K-slice
    // fragment gather (documented 16-bit A/C layouts; B symmetric, col-per-lane)
    const unsigned off = (lane < 16) ? 0u : 4u;
    const unsigned* a32 = (const unsigned*)&lA[cur][(wm * 16 + (lane & 15)) * 32];
    const unsigned* b32a = (const unsigned*)&lB[cur][(wn * 32 + (lane & 15)) * 32];
    const unsigned* b32b = (const unsigned*)&lB[cur][(wn * 32 + 16 + (lane & 15)) * 32];
    union { unsigned u[8]; v16h h; } fa, fb0, fb1;
#pragma unroll
    for (int i = 0; i < 4; ++i) {
      fa.u[i]      = a32[off + i];
      fa.u[4 + i]  = a32[off + 8 + i];
      fb0.u[i]     = b32a[off + i];
      fb0.u[4 + i] = b32a[off + 8 + i];
      fb1.u[i]     = b32b[off + i];
      fb1.u[4 + i] = b32b[off + 8 + i];
    }
    c0 = __builtin_amdgcn_wmma_f32_16x16x32_f16(false, fa.h, false, fb0.h, (short)0, c0,
                                                false, false);
    c1 = __builtin_amdgcn_wmma_f32_16x16x32_f16(false, fa.h, false, fb1.h, (short)0, c1,
                                                false, false);
    if (kt + 1 < NT) asm volatile("s_wait_asynccnt 0x0" ::: "memory");
    __syncthreads();
  }
  // epilogue: elem e -> row = e (+8 for lanes>=16), col = lane%16
  unsigned colA = col0 + wn * 32 + (lane & 15);
  unsigned colB = colA + 16;
  unsigned rbase = row0 + wm * 16 + ((lane < 16) ? 0u : 8u);
  float bv0 = bias[colA], bv1 = bias[colB];
#pragma unroll
  for (int e = 0; e < 8; ++e) {
    float v0 = c0[e] + bv0;
    float v1 = c1[e] + bv1;
    if constexpr (OP == 1) { v0 = v0 > 0.0f ? v0 : 0.0f; v1 = v1 > 0.0f ? v1 : 0.0f; }
    if constexpr (OP == 2) {
      v0 = v0 > 0.0f ? v0 + 1.0f : expf(v0);  // elu(x)+1
      v1 = v1 > 0.0f ? v1 + 1.0f : expf(v1);
    }
    size_t i0 = (size_t)(rbase + e) * N + colA;
    size_t i1 = (size_t)(rbase + e) * N + colB;
    if constexpr (OUTM == 0 || OUTM == 2) { Cf[i0] = v0; Cf[i1] = v1; }
    if constexpr (OUTM == 1 || OUTM == 2) { Ch[i0] = (f16)v0; Ch[i1] = (f16)v1; }
  }
}

// ---------------------------------------------------------------------------
// Linear attention: ctx[bs,h] = sum_s phiK[s]^T outer V[s]; phiKsum[bs,h]
// ---------------------------------------------------------------------------
__global__ void k_ctx(const float* __restrict__ phiK, const float* __restrict__ V,
                      float* __restrict__ ctx, float* __restrict__ ksum) {
  __shared__ float sK[4][64], sV[4][64];
  const int bh = blockIdx.x;
  const int bs = bh >> 2, h = bh & 3;
  const int t = threadIdx.x;
  const int i = t >> 2, jb = (t & 3) << 4;
  float acc[16];
#pragma unroll
  for (int jj = 0; jj < 16; ++jj) acc[jj] = 0.0f;
  float kacc = 0.0f;
  for (int s0 = 0; s0 < PP; s0 += 4) {
    for (int e = t; e < 512; e += 256) {
      int rowi = e >> 7, col = e & 127;
      size_t base = ((size_t)(s0 + rowi) * BS + bs) * DD + h * 64;
      if (col < 64) sK[rowi][col] = phiK[base + col];
      else          sV[rowi][col - 64] = V[base + (col - 64)];
    }
    __syncthreads();
#pragma unroll
    for (int ss = 0; ss < 4; ++ss) {
      float kv = sK[ss][i];
      if ((t & 3) == 0) kacc += kv;
#pragma unroll
      for (int jj = 0; jj < 16; ++jj) acc[jj] += kv * sV[ss][jb + jj];
    }
    __syncthreads();
  }
  size_t cbase = ((size_t)bh * 64 + i) * 64 + jb;
#pragma unroll
  for (int jj = 0; jj < 16; ++jj) ctx[cbase + jj] = acc[jj];
  if ((t & 3) == 0) ksum[(size_t)bh * 64 + i] = kacc;
}

// ---------------------------------------------------------------------------
// out[s,bs,h,:] = (phiQ[s] @ ctx) / (phiQ[s].phiKsum + eps) -> f16 attn buffer
// ---------------------------------------------------------------------------
__global__ void k_attn(const float* __restrict__ phiQ, const float* __restrict__ ctx,
                       const float* __restrict__ ksum, f16* __restrict__ attn) {
  __shared__ float sCtx[64 * 64];
  __shared__ float sKs[64];
  __shared__ float sDen[PP];
  const int bh = blockIdx.x;
  const int bs = bh >> 2, h = bh & 3;
  const int t = threadIdx.x;
  size_t cbase = (size_t)bh * 4096;
  for (int e = t; e < 4096; e += 256) sCtx[e] = ctx[cbase + e];
  if (t < 64) sKs[t] = ksum[(size_t)bh * 64 + t];
  __syncthreads();
  if (t < PP) {
    const float* q = phiQ + ((size_t)t * BS + bs) * DD + h * 64;
    float d = 0.0f;
#pragma unroll
    for (int i = 0; i < 64; ++i) d += q[i] * sKs[i];
    sDen[t] = d + EPSF;
  }
  __syncthreads();
  const int j = t & 63;
  for (int sb = 0; sb < PP; sb += 4) {
    int s = sb + (t >> 6);
    const float* q = phiQ + ((size_t)s * BS + bs) * DD + h * 64;
    float a = 0.0f;
#pragma unroll
    for (int i = 0; i < 64; ++i) a += q[i] * sCtx[i * 64 + j];
    attn[((size_t)s * BS + bs) * DD + h * 64 + j] = (f16)(a / sDen[s]);
  }
}

// ---------------------------------------------------------------------------
// Residual + LayerNorm: y = LN(x0 + x1) * g + be ; writes f32 + f16
// ---------------------------------------------------------------------------
__global__ void k_ln(const float* __restrict__ x0, const float* __restrict__ x1,
                     const float* __restrict__ g, const float* __restrict__ be,
                     float* __restrict__ of, f16* __restrict__ oh) {
  __shared__ float red[256];
  const int r = blockIdx.x, t = threadIdx.x;
  size_t idx = (size_t)r * DD + t;
  float x = x0[idx] + x1[idx];
  red[t] = x;
  __syncthreads();
  for (int s = 128; s > 0; s >>= 1) { if (t < s) red[t] += red[t + s]; __syncthreads(); }
  float mean = red[0] * (1.0f / DD);
  __syncthreads();
  float d = x - mean;
  red[t] = d * d;
  __syncthreads();
  for (int s = 128; s > 0; s >>= 1) { if (t < s) red[t] += red[t + s]; __syncthreads(); }
  float var = red[0] * (1.0f / DD);
  float y = d * rsqrtf(var + 1e-5f) * g[t] + be[t];
  of[idx] = y;
  oh[idx] = (f16)y;
}

// ---------------------------------------------------------------------------
// Final outputs
// ---------------------------------------------------------------------------
__global__ void k_feats(const float* __restrict__ act, float* __restrict__ out) {
  const int blk = blockIdx.x, t = threadIdx.x;
  int b = blk / (SS * PP), sp = blk % (SS * PP);
  int s = sp / PP, p = sp % PP;
  size_t row = (size_t)p * BS + b * SS + s;
  out[(size_t)blk * DD + t] = act[row * DD + t];
}

__global__ void k_xyz(const float* __restrict__ edges, float* __restrict__ out) {
  int id = blockIdx.x * 256 + threadIdx.x;
  if (id >= BB * SS * PP) return;
  int b = id / (SS * PP), sp = id % (SS * PP);
  int s = sp / PP, p = sp % PP;
  int iy = p / 10, ix = p % 10;
  float gx = (ix * 16 + 8.0f) / (float)(WW - 1);
  float gy = (iy * 16 + 8.0f) / (float)(HH - 1);
  float* o = out + (size_t)MROWS * DD + (size_t)id * 3;
  o[0] = -1.0f + gx * (2.0f + EPSF);
  o[1] = -1.0f + gy * (2.0f + EPSF);
  o[2] = 0.5f * (edges[b * (SS + 1) + s] + edges[b * (SS + 1) + s + 1]);
}

// ---------------------------------------------------------------------------
// Host driver
// ---------------------------------------------------------------------------
extern "C" void kernel_launch(void* const* d_in, const int* in_sizes, int n_in,
                              void* d_out, int out_size, void* d_ws, size_t ws_size,
                              hipStream_t stream) {
  (void)in_sizes; (void)n_in; (void)out_size; (void)ws_size;
  const float* xyz     = (const float*)d_in[0];
  const float* patch_w = (const float*)d_in[1];
  const float* patch_b = (const float*)d_in[2];
  struct Layer {
    const float *wq, *bq, *wk, *bk, *wv, *bv, *wo, *bo, *g1, *be1, *w1, *fb1, *w2, *fb2, *g2, *be2;
  } L[4];
  for (int l = 0; l < 4; ++l) {
    int base = 3 + 16 * l;
    L[l].wq  = (const float*)d_in[base + 0];  L[l].bq  = (const float*)d_in[base + 1];
    L[l].wk  = (const float*)d_in[base + 2];  L[l].bk  = (const float*)d_in[base + 3];
    L[l].wv  = (const float*)d_in[base + 4];  L[l].bv  = (const float*)d_in[base + 5];
    L[l].wo  = (const float*)d_in[base + 6];  L[l].bo  = (const float*)d_in[base + 7];
    L[l].g1  = (const float*)d_in[base + 8];  L[l].be1 = (const float*)d_in[base + 9];
    L[l].w1  = (const float*)d_in[base + 10]; L[l].fb1 = (const float*)d_in[base + 11];
    L[l].w2  = (const float*)d_in[base + 12]; L[l].fb2 = (const float*)d_in[base + 13];
    L[l].g2  = (const float*)d_in[base + 14]; L[l].be2 = (const float*)d_in[base + 15];
  }

  // workspace carve-up (deterministic)
  char* ws = (char*)d_ws;
  size_t off = 0;
  auto take = [&](size_t bytes) -> char* {
    char* p = ws + off;
    off += (bytes + 255) & ~(size_t)255;
    return p;
  };
  unsigned* counts = (unsigned*)take((size_t)BB * SS * HH * WW * 4);  // reused as bevlog f32
  float*    edges  = (float*)take(BB * (SS + 1) * 4);
  float*    sMn    = (float*)take(BS * 4);
  float*    sMx    = (float*)take(BS * 4);
  f16*      X16    = (f16*)take((size_t)MROWS * DD * 2);
  f16*      pw16   = (f16*)take((size_t)DD * DD * 2);
  f16 *wq16[4], *wk16[4], *wv16[4], *wo16[4], *w116[4], *w216[4];
  for (int l = 0; l < 4; ++l) {
    wq16[l] = (f16*)take((size_t)DD * DD * 2);
    wk16[l] = (f16*)take((size_t)DD * DD * 2);
    wv16[l] = (f16*)take((size_t)DD * DD * 2);
    wo16[l] = (f16*)take((size_t)DD * DD * 2);
    w116[l] = (f16*)take((size_t)DD * 2 * DD * 2);
    w216[l] = (f16*)take((size_t)2 * DD * DD * 2);
  }
  float* actA   = (float*)take((size_t)MROWS * DD * 4);
  float* actB   = (float*)take((size_t)MROWS * DD * 4);
  f16*   actA16 = (f16*)take((size_t)MROWS * DD * 2);
  f16*   actB16 = (f16*)take((size_t)MROWS * DD * 2);
  float* Qb     = (float*)take((size_t)MROWS * DD * 4);
  float* Kb     = (float*)take((size_t)MROWS * DD * 4);
  float* Vb     = (float*)take((size_t)MROWS * DD * 4);
  f16*   attn16 = (f16*)take((size_t)MROWS * DD * 2);
  float* proj   = (float*)take((size_t)MROWS * DD * 4);
  f16*   hid16  = (f16*)take((size_t)MROWS * 2 * DD * 2);
  float* ctxb   = (float*)take((size_t)BS * 4 * 64 * 64 * 4);
  float* ksum   = (float*)take((size_t)BS * 4 * 64 * 4);

  float* out = (float*)d_out;

  // ---- BEV generation ----
  hipMemsetAsync(counts, 0, (size_t)BB * SS * HH * WW * 4, stream);
  k_zminmax<<<BB, 256, 0, stream>>>(xyz, edges);
  k_scatter<<<(int)(((size_t)BB * NN + 255) / 256), 256, 0, stream>>>(xyz, edges, counts);
  k_bevlog<<<BS, 256, 0, stream>>>(counts, sMn, sMx);
  k_patchify<<<MROWS, 256, 0, stream>>>((const float*)counts, sMn, sMx, X16);

  // ---- weight conversion to f16 (transposed: [N][K]) ----
  k_cvtT<<<(DD * DD + 255) / 256, 256, 0, stream>>>(patch_w, pw16, DD, DD);
  for (int l = 0; l < 4; ++l) {
    k_cvtT<<<(DD * DD + 255) / 256, 256, 0, stream>>>(L[l].wq, wq16[l], DD, DD);
    k_cvtT<<<(DD * DD + 255) / 256, 256, 0, stream>>>(L[l].wk, wk16[l], DD, DD);
    k_cvtT<<<(DD * DD + 255) / 256, 256, 0, stream>>>(L[l].wv, wv16[l], DD, DD);
    k_cvtT<<<(DD * DD + 255) / 256, 256, 0, stream>>>(L[l].wo, wo16[l], DD, DD);
    k_cvtT<<<(DD * 2 * DD + 255) / 256, 256, 0, stream>>>(L[l].w1, w116[l], DD, 2 * DD);
    k_cvtT<<<(2 * DD * DD + 255) / 256, 256, 0, stream>>>(L[l].w2, w216[l], 2 * DD, DD);
  }

  // ---- patch embedding: actA = X @ patch_w + b (f32 + f16) ----
  dim3 g256(MROWS / 64, DD / 64), g512(MROWS / 64, (2 * DD) / 64);
  k_gemm<DD, DD, 2, 0><<<g256, 256, 0, stream>>>(X16, pw16, patch_b, actA, actA16);

  // ---- encoder layers ----
  for (int l = 0; l < 4; ++l) {
    // QKV (phi fused into Q,K epilogue)
    k_gemm<DD, DD, 0, 2><<<g256, 256, 0, stream>>>(actA16, wq16[l], L[l].bq, Qb, nullptr);
    k_gemm<DD, DD, 0, 2><<<g256, 256, 0, stream>>>(actA16, wk16[l], L[l].bk, Kb, nullptr);
    k_gemm<DD, DD, 0, 0><<<g256, 256, 0, stream>>>(actA16, wv16[l], L[l].bv, Vb, nullptr);
    // linear attention
    k_ctx<<<BS * 4, 256, 0, stream>>>(Kb, Vb, ctxb, ksum);
    k_attn<<<BS * 4, 256, 0, stream>>>(Qb, ctxb, ksum, attn16);
    // output projection + LN1
    k_gemm<DD, DD, 0, 0><<<g256, 256, 0, stream>>>(attn16, wo16[l], L[l].bo, proj, nullptr);
    k_ln<<<MROWS, 256, 0, stream>>>(actA, proj, L[l].g1, L[l].be1, actB, actB16);
    // FFN + LN2
    k_gemm<DD, 2 * DD, 1, 1><<<g512, 256, 0, stream>>>(actB16, w116[l], L[l].fb1, nullptr, hid16);
    k_gemm<2 * DD, DD, 0, 0><<<g256, 256, 0, stream>>>(hid16, w216[l], L[l].fb2, proj, nullptr);
    k_ln<<<MROWS, 256, 0, stream>>>(actB, proj, L[l].g2, L[l].be2, actA, actA16);
  }

  // ---- outputs ----
  k_feats<<<MROWS, 256, 0, stream>>>(actA, out);
  k_xyz<<<(BB * SS * PP + 255) / 256, 256, 0, stream>>>(edges, out);
}
